// EnhancedCrossAttention_42726334661045
// MI455X (gfx1250) — compile-verified, matrix operand order checked by reading, not run on hardware
//
#include <hip/hip_runtime.h>

// ---------------------------------------------------------------------------
// EnhancedCrossAttention for MI455X (gfx1250, wave32, WMMA bf16)
//   B=2048, T=32, D=2048, H=16, hd=128
// Memory-bound (~2.2 GB total HBM traffic @ 23.3 TB/s => ~95us floor):
//   - bf16 WMMA (v_wmma_f32_16x16x32_bf16) for the four big GEMMs
//   - K/V materialized once as bf16 in natural (T,B,D) layout (coalesced)
//   - double-buffered LDS pipeline: global loads of tile i+1 overlap WMMA of i
//   - all 8 B-fragments preloaded before the 8 WMMAs (no per-WMMA dscnt-0)
//   - tiny attention core (268 MFLOP) in VALU f32 with wave32 shuffles
// Workspace (d_ws): [Q/O bf16 8MB][K bf16 256MB][V bf16 256MB] ~= 520MB
// ---------------------------------------------------------------------------

typedef unsigned short ushort_t;
typedef __attribute__((ext_vector_type(16))) __bf16 v16bf;
typedef __attribute__((ext_vector_type(8)))  float  v8f;

#define BDIM 2048   // batch
#define TDIM 32     // time steps
#define DDIM 2048   // model dim (= N = K of all GEMMs)
#define HN   16     // heads
#define HD   128    // head dim
#define LDA  40     // padded LDS row stride (bf16 elems) -> 80B, bank-spread

__device__ __forceinline__ unsigned pk2bf(float lo, float hi) {
  // two f32 -> packed bf16x2 with round-to-nearest-even
  unsigned a = __float_as_uint(lo), b = __float_as_uint(hi);
  a += 0x7FFFu + ((a >> 16) & 1u);
  b += 0x7FFFu + ((b >> 16) & 1u);
  return (a >> 16) | (b & 0xFFFF0000u);
}
__device__ __forceinline__ ushort_t f2bf(float f) {
  unsigned u = __float_as_uint(f);
  u += 0x7FFFu + ((u >> 16) & 1u);
  return (ushort_t)(u >> 16);
}
__device__ __forceinline__ float bf2f(ushort_t s) {
  return __uint_as_float(((unsigned)s) << 16);
}

union FragAB { v16bf v; uint4 q[2]; };
union Pack16 { unsigned w[8]; uint4 q[2]; };
union Pack8  { ushort_t u[8]; uint4 q; };
union Pack4  { ushort_t u[4]; uint2 q; };

// ---------------------------------------------------------------------------
// Tiled GEMM:  out[m,n] = sum_k A[m,k] * W[n,k] + bias[n]
//   block tile 128(M) x 128(N), K-step 32, 8 waves; wave w -> rows 16w..16w+15
//   double-buffered LDS, one __syncthreads per K-step.
//   A_BF16  : A is bf16 (final projection reads attention output)
//   ADD_POS : A[m,k] += clip(pos[m>>11][k], -10, 10)   (keys projection)
//   OUT_F32 : write f32 (final output), else bf16 row-major
// ---------------------------------------------------------------------------
template<bool A_BF16, bool ADD_POS, bool OUT_F32>
__global__ __launch_bounds__(256)
void gemm_wmma_kernel(const void* __restrict__ Aptr, const float* __restrict__ W,
                      const float* __restrict__ bias, const float* __restrict__ pos,
                      void* __restrict__ outp)
{
  __shared__ __align__(16) ushort_t lds_a[2][128 * LDA];
  __shared__ __align__(16) ushort_t lds_b[2][128 * LDA];

  const int nb   = blockIdx.x, mb = blockIdx.y;
  const int tid  = threadIdx.x;
  const int wave = tid >> 5, lane = tid & 31;

  // cooperative tile-load mapping: 256 threads -> 128 rows x 2 segments of 16
  const int lrow = tid >> 1, lseg = (tid & 1) * 16;
  const size_t gmA = (size_t)mb * 128 + lrow;
  const size_t gnB = (size_t)nb * 128 + lrow;

  // fragment mapping (ISA 7.12.2, 16-bit 16x32 layout):
  //   lane<16: K chunks {0..7,16..23}; lane>=16: {8..15,24..31}
  const int mrow = (wave << 4) + (lane & 15);
  const int cc   = (lane >> 4) << 3;          // 0 or 8

  auto loadA = [&](int kk, Pack16& r) {
    if (A_BF16) {
      const ushort_t* s = (const ushort_t*)Aptr + gmA * DDIM + kk + lseg;
      r.q[0] = ((const uint4*)s)[0];
      r.q[1] = ((const uint4*)s)[1];
      if (kk + 64 < DDIM) __builtin_prefetch(s + 64, 0, 1);
    } else {
      const float* s = (const float*)Aptr + gmA * DDIM + kk + lseg;
      if (ADD_POS) {
        const float* pp = pos + (gmA >> 11) * DDIM + kk + lseg;  // t = m / 2048
        #pragma unroll
        for (int j = 0; j < 8; j++) {
          const float a0 = s[2*j]     + fminf(10.f, fmaxf(-10.f, pp[2*j]));
          const float a1 = s[2*j + 1] + fminf(10.f, fmaxf(-10.f, pp[2*j + 1]));
          r.w[j] = pk2bf(a0, a1);
        }
      } else {
        #pragma unroll
        for (int j = 0; j < 8; j++) r.w[j] = pk2bf(s[2*j], s[2*j + 1]);
      }
      if (kk + 64 < DDIM) __builtin_prefetch(s + 64, 0, 1);
    }
  };
  auto loadB = [&](int kk, Pack16& r) {
    const float* s = W + gnB * DDIM + kk + lseg;
    #pragma unroll
    for (int j = 0; j < 8; j++) r.w[j] = pk2bf(s[2*j], s[2*j + 1]);
    if (kk + 64 < DDIM) __builtin_prefetch(s + 64, 0, 1);
  };

  v8f acc[8] = {};
  Pack16 ra, rb;
  loadA(0, ra);
  loadB(0, rb);

  int buf = 0;
  for (int kk = 0; kk < DDIM; kk += 32) {
    // stage tile kk into LDS[buf] (nobody reads this buffer right now)
    *(uint4*)&lds_a[buf][lrow * LDA + lseg]     = ra.q[0];
    *(uint4*)&lds_a[buf][lrow * LDA + lseg + 8] = ra.q[1];
    *(uint4*)&lds_b[buf][lrow * LDA + lseg]     = rb.q[0];
    *(uint4*)&lds_b[buf][lrow * LDA + lseg + 8] = rb.q[1];

    // kick off global loads for tile kk+32 (in flight during the WMMAs)
    if (kk + 32 < DDIM) {
      loadA(kk + 32, ra);
      loadB(kk + 32, rb);
    }

    __syncthreads();   // LDS[buf] visible; prev iter's reads of LDS[buf^1] done

    // preload A fragment + all 8 B fragments, then 8 back-to-back WMMAs
    FragAB a;
    a.q[0] = *(const uint4*)&lds_a[buf][mrow * LDA + cc];
    a.q[1] = *(const uint4*)&lds_a[buf][mrow * LDA + cc + 16];
    FragAB bf[8];
    #pragma unroll
    for (int j = 0; j < 8; j++) {
      const int nrow = (j << 4) + (lane & 15);
      bf[j].q[0] = *(const uint4*)&lds_b[buf][nrow * LDA + cc];
      bf[j].q[1] = *(const uint4*)&lds_b[buf][nrow * LDA + cc + 16];
    }
    #pragma unroll
    for (int j = 0; j < 8; j++) {
      acc[j] = __builtin_amdgcn_wmma_f32_16x16x32_bf16(
          false, a.v, false, bf[j].v, (short)0, acc[j], false, false);
    }
    buf ^= 1;
  }

  // epilogue: C layout lane -> (N = lane&15, M = r + 8*(lane>=16))
  const int nlo = lane & 15;
  const int mhi = (lane >> 4) << 3;
  #pragma unroll
  for (int j = 0; j < 8; j++) {
    const int gn = nb * 128 + (j << 4) + nlo;
    const float bb = bias[gn];
    #pragma unroll
    for (int r = 0; r < 8; r++) {
      const size_t gm = (size_t)mb * 128 + (wave << 4) + mhi + r;
      const float val = acc[j][r] + bb;
      if (OUT_F32) ((float*)outp)[gm * DDIM + gn] = val;
      else         ((ushort_t*)outp)[gm * DDIM + gn] = f2bf(val);
    }
  }
}

// ---------------------------------------------------------------------------
// Attention core: one block per batch row b, one wave32 per head h.
// K/V are bf16 in natural (T, B, D) layout. lane t computes score(b,h,t);
// softmax via wave32 shuffles; lanes then accumulate 4 output dims each.
// O (bf16) aliases Q's ws region: Q[b] is fully staged to LDS before any
// O[b] write, and different blocks touch disjoint b slices -> safe.
// ---------------------------------------------------------------------------
__global__ __launch_bounds__(512)
void attention_kernel(const ushort_t* __restrict__ Qbf, const ushort_t* __restrict__ Kbf,
                      const ushort_t* __restrict__ Vbf, const unsigned char* __restrict__ mask,
                      ushort_t* __restrict__ Obf)
{
  __shared__ float q_sh[HN * HD];
  __shared__ float p_sh[HN * TDIM];
  const int b = blockIdx.x;
  const int tid = threadIdx.x;          // 512 = 16 waves
  const int h = tid >> 5, lane = tid & 31;

  { // stage q for all heads
    Pack4 r; r.q = *(const uint2*)&Qbf[(size_t)b * DDIM + tid * 4];
    #pragma unroll
    for (int i = 0; i < 4; i++) q_sh[tid * 4 + i] = bf2f(r.u[i]);
  }
  __syncthreads();

  // score for (b, h, t=lane): K row at (t, b, h*HD .. ) is 256B contiguous
  const ushort_t* krow = Kbf + ((size_t)lane * BDIM + b) * DDIM + h * HD;
  float s = 0.f;
  #pragma unroll
  for (int d8 = 0; d8 < HD; d8 += 8) {
    Pack8 kr; kr.q = *(const uint4*)&krow[d8];
    #pragma unroll
    for (int i = 0; i < 8; i++) s += q_sh[h * HD + d8 + i] * bf2f(kr.u[i]);
  }
  s *= 0.08838834764831845f;            // 1/sqrt(128)
  s = fminf(50.f, fmaxf(-50.f, s));
  if (!mask[b * TDIM + lane]) s = -1e30f;

  float m = s;
  #pragma unroll
  for (int off = 16; off > 0; off >>= 1) m = fmaxf(m, __shfl_xor(m, off, 32));
  const float p = __expf(s - m);
  float l = p;
  #pragma unroll
  for (int off = 16; off > 0; off >>= 1) l += __shfl_xor(l, off, 32);
  p_sh[h * TDIM + lane] = p / l;
  __syncthreads();

  // out[b,h,d] = sum_t attn * v ; lane handles 4 dims
  float acc4[4] = {0.f, 0.f, 0.f, 0.f};
  #pragma unroll 4
  for (int t = 0; t < TDIM; t++) {
    const float w = p_sh[h * TDIM + t];
    const ushort_t* vrow = Vbf + ((size_t)t * BDIM + b) * DDIM + h * HD + lane * 4;
    Pack4 vr; vr.q = *(const uint2*)vrow;
    #pragma unroll
    for (int i = 0; i < 4; i++) acc4[i] += w * bf2f(vr.u[i]);
  }
  Pack4 o;
  #pragma unroll
  for (int i = 0; i < 4; i++) o.u[i] = f2bf(acc4[i]);
  *(uint2*)&Obf[(size_t)b * DDIM + h * HD + lane * 4] = o.q;
}

// ---------------------------------------------------------------------------
extern "C" void kernel_launch(void* const* d_in, const int* in_sizes, int n_in,
                              void* d_out, int out_size, void* d_ws, size_t ws_size,
                              hipStream_t stream)
{
  (void)in_sizes; (void)n_in; (void)out_size; (void)ws_size;

  const float* query  = (const float*)d_in[0];
  const float* keys   = (const float*)d_in[1];
  const float* values = (const float*)d_in[2];
  const unsigned char* mask = (const unsigned char*)d_in[3];
  const float* pos = (const float*)d_in[4];
  const float* Wq = (const float*)d_in[5];
  const float* bq = (const float*)d_in[6];
  const float* Wk = (const float*)d_in[7];
  const float* bk = (const float*)d_in[8];
  const float* Wv = (const float*)d_in[9];
  const float* bv = (const float*)d_in[10];
  const float* Wo = (const float*)d_in[11];
  const float* bo = (const float*)d_in[12];
  float* out = (float*)d_out;

  const size_t QO_BYTES = (size_t)BDIM * DDIM * 2;          // 8 MB
  const size_t KV_BYTES = (size_t)TDIM * BDIM * DDIM * 2;   // 256 MB
  char* ws = (char*)d_ws;                                   // needs ~520 MB
  ushort_t* Qbf = (ushort_t*)(ws);                          // also reused as O
  ushort_t* Kbf = (ushort_t*)(ws + QO_BYTES);               // (T,B,D) bf16
  ushort_t* Vbf = (ushort_t*)(ws + QO_BYTES + KV_BYTES);    // (T,B,D) bf16

  const dim3 blk(256);

  // 1) Q projection: M = 2048
  gemm_wmma_kernel<false, false, false>
      <<<dim3(16, BDIM / 128), blk, 0, stream>>>(query, Wq, bq, nullptr, Qbf);

  // 2) K projection (pos add + clip): M = T*B = 65536, out (T,B,D) bf16
  gemm_wmma_kernel<false, true, false>
      <<<dim3(16, (TDIM * BDIM) / 128), blk, 0, stream>>>(keys, Wk, bk, pos, Kbf);

  // 3) V projection: out (T,B,D) bf16
  gemm_wmma_kernel<false, false, false>
      <<<dim3(16, (TDIM * BDIM) / 128), blk, 0, stream>>>(values, Wv, bv, nullptr, Vbf);

  // 4) attention core (writes O over Q's region — safe, see kernel comment)
  attention_kernel<<<dim3(BDIM), dim3(512), 0, stream>>>(Qbf, Kbf, Vbf, mask, Qbf);

  // 5) output projection: M = 2048, f32 output
  gemm_wmma_kernel<true, false, true>
      <<<dim3(16, BDIM / 128), blk, 0, stream>>>(Qbf, Wo, bo, nullptr, out);
}
